// SelfAttention_69793218560643
// MI455X (gfx1250) — matrix operand order repaired and checked by reading, not compiled, not run on hardware
//
#include <hip/hip_runtime.h>
#include <hip/hip_bf16.h>
#include <stdint.h>

#define N_B 4
#define L_S 1024
#define E_D 1024
#define H_N 32
#define HD  32

typedef __attribute__((ext_vector_type(16))) __bf16 v16bf;
typedef __attribute__((ext_vector_type(8)))  float  v8f;

union BFrag { v16bf v; __bf16 e[16]; };

__device__ __forceinline__ uint32_t pack_bf2(float a, float b) {
    union { __bf16 h[2]; uint32_t u; } p;
    p.h[0] = (__bf16)a;
    p.h[1] = (__bf16)b;
    return p.u;
}

// Async global->LDS 16-byte copy (per-lane), tracked by ASYNCcnt.
__device__ __forceinline__ void async_b128(uint32_t lds_addr, uint64_t gaddr) {
    asm volatile("global_load_async_to_lds_b128 %0, %1, off"
                 :: "v"(lds_addr), "v"(gaddr) : "memory");
}
__device__ __forceinline__ void wait_async0() {
    asm volatile("s_wait_asynccnt 0" ::: "memory");
}
__device__ __forceinline__ uint32_t lds_off(const void* p) {
    return (uint32_t)(uintptr_t)p;    // low 32 bits of shared-aperture address
}

// A-matrix 16x32 bf16 fragment: lane<16 -> row=lane, K={0..15};
// lane>=16 -> row=lane-16, K={16..31}
__device__ __forceinline__ v16bf gather_a(const __bf16* base, int ld, int lane) {
    BFrag f;
    int m  = lane & 15;
    int kb = (lane & 16) ? 16 : 0;
#pragma unroll
    for (int i = 0; i < 16; ++i) {
        int k = kb + i;
        f.e[i] = base[m * ld + k];
    }
    return f.v;
}

// B-matrix 32x16 bf16 fragment where element(col,k) = base[(n0+col)*ld + k]
__device__ __forceinline__ v16bf gather_b_row(const __bf16* base, int ld, int n0, int lane) {
    BFrag f;
    int col = (lane & 15) + n0;
    int kb  = (lane & 16) ? 16 : 0;
#pragma unroll
    for (int i = 0; i < 16; ++i) f.e[i] = base[col * ld + kb + i];
    return f.v;
}

__device__ __forceinline__ v8f wmma_bf16(v16bf a, v16bf b, v8f c) {
    return __builtin_amdgcn_wmma_f32_16x16x32_bf16(false, a, false, b, (short)0, c, false, false);
}

// ---------------------------------------------------------------------------
// Kernel 1: per-head projection  Y[l][d] = sum_c X[l][h*32+c] * W[d][c]
// grid: (N*L/128, H), block 256.  wscale folds 1/sqrt(E) into Wq.
// ---------------------------------------------------------------------------
__global__ __launch_bounds__(256) void proj_kernel(
    const float* __restrict__ x, const float* __restrict__ w,
    __bf16* __restrict__ out, float wscale) {
    __shared__ __bf16 Xl[128 * HD];
    __shared__ __bf16 Wl[HD * HD];
    const int tid  = threadIdx.x;
    const int lane = tid & 31;
    const int wave = tid >> 5;
    const int l0   = blockIdx.x * 128;
    const int n    = l0 / L_S;
    const int lrow = l0 % L_S;
    const int h    = blockIdx.y;

    // stage X tile (128x32) f32 -> bf16, packed pair stores
    uint32_t* xd = (uint32_t*)Xl;
#pragma unroll
    for (int i = 0; i < 8; ++i) {
        int d = tid * 8 + i;                 // dword index, 2048 total
        int r = d >> 4, c = (d & 15) * 2;
        const float* src = &x[(size_t)(n * L_S + lrow + r) * E_D + h * HD + c];
        xd[d] = pack_bf2(src[0], src[1]);
    }
    uint32_t* wd = (uint32_t*)Wl;
#pragma unroll
    for (int i = 0; i < 2; ++i) {
        int d = tid * 2 + i;                 // 512 dwords
        wd[d] = pack_bf2(w[d * 2] * wscale, w[d * 2 + 1] * wscale);
    }
    __syncthreads();

    v16bf a  = gather_a(Xl + wave * 16 * HD, HD, lane);
    v16bf b0 = gather_b_row(Wl, HD, 0, lane);
    v16bf b1 = gather_b_row(Wl, HD, 16, lane);
    v8f z = {};
    v8f c0 = wmma_bf16(a, b0, z);
    v8f c1 = wmma_bf16(a, b1, z);

    __bf16* ob = out + ((size_t)(n * H_N + h) * L_S + lrow + wave * 16) * HD;
    int rbase = (lane & 16) ? 8 : 0;
    int col   = lane & 15;
#pragma unroll
    for (int j = 0; j < 8; ++j) {
        ob[(size_t)(rbase + j) * HD + col]      = (__bf16)c0[j];
        ob[(size_t)(rbase + j) * HD + 16 + col] = (__bf16)c1[j];
    }
}

// ---------------------------------------------------------------------------
// Kernel 2: flash attention per (n, h, 128 q-rows). block 256 = 8 waves,
// each wave owns 16 q rows. K/V streamed in 32-row chunks. K + Q tiles are
// staged via async global->LDS b128; V is staged transposed by waves 4-7.
// ---------------------------------------------------------------------------
__global__ __launch_bounds__(256) void attn_kernel(
    const __bf16* __restrict__ Qp, const __bf16* __restrict__ Kp,
    const __bf16* __restrict__ Vp, const int* __restrict__ mask,
    __bf16* __restrict__ AO) {
    __shared__ __bf16 Qls[128 * HD];
    __shared__ __bf16 Kl[32 * HD];       // [k][d]
    __shared__ __bf16 Vt[HD * 32];       // transposed: [d][k]
    __shared__ __bf16 Pl[8][16 * 32];

    const int tid = threadIdx.x, lane = tid & 31, wave = tid >> 5;
    const int q0 = blockIdx.x * 128, h = blockIdx.y, n = blockIdx.z;
    const size_t headbase = (size_t)(n * H_N + h) * L_S * HD;

    {   // async stage Q tile: 8KB = 512 x b128 slots, 2 per thread
        uint64_t gq = (uint64_t)(uintptr_t)(Qp + headbase + (size_t)q0 * HD);
        uint32_t lq = lds_off(Qls);
#pragma unroll
        for (int i = 0; i < 2; ++i) {
            int s = tid * 2 + i;
            async_b128(lq + s * 16, gq + (uint32_t)(s * 16));
        }
        wait_async0();
    }
    __syncthreads();
    v16bf qf = gather_a(Qls + wave * 16 * HD, HD, lane);

    v8f o0 = {}; v8f o1 = {};
    float m[8], lsum[8];
#pragma unroll
    for (int j = 0; j < 8; ++j) { m[j] = -3.4e38f; lsum[j] = 0.f; }
    const int col   = lane & 15;
    const int rbase = (lane & 16) ? 8 : 0;
    const uint32_t lk = lds_off(Kl);

    for (int kc = 0; kc < L_S; kc += 32) {
        __syncthreads();                    // previous chunk fully consumed
        if (wave < 4) {
            // waves 0-3: async K chunk (2KB = 128 b128 slots, 1/thread)
            uint64_t gk = (uint64_t)(uintptr_t)(Kp + headbase + (size_t)kc * HD);
            async_b128(lk + tid * 16, gk + (uint32_t)(tid * 16));
        } else {
            // waves 4-7: stage V chunk transposed (Vt[d][k] = V[k][d])
            const __bf16* gv = Vp + headbase + (size_t)kc * HD;
            int t = tid - 128;
#pragma unroll
            for (int i = 0; i < 8; ++i) {
                int e = t * 8 + i;          // 1024 elems; k = e>>5 fixed/thread
                int k = e >> 5, d = e & 31;
                Vt[d * 32 + k] = gv[e];
            }
        }
        wait_async0();
        __syncthreads();

        // S = Q (16x32) @ K_chunk^T  -> 16x32 scores (pre-scaled via Wq)
        v16bf bk0 = gather_b_row(Kl, HD, 0, lane);
        v16bf bk1 = gather_b_row(Kl, HD, 16, lane);
        v8f z = {};
        v8f s0 = wmma_bf16(qf, bk0, z);
        v8f s1 = wmma_bf16(qf, bk1, z);

        int mv0 = mask[n * L_S + kc + col];
        int mv1 = mask[n * L_S + kc + 16 + col];
        if (mv0 == 0) {
#pragma unroll
            for (int j = 0; j < 8; ++j) s0[j] = -3.0e18f;
        }
        if (mv1 == 0) {
#pragma unroll
            for (int j = 0; j < 8; ++j) s1[j] = -3.0e18f;
        }

        // online softmax: per-row chunk max (rows live in 16-lane halves)
        float cm[8];
#pragma unroll
        for (int j = 0; j < 8; ++j) cm[j] = fmaxf(s0[j], s1[j]);
#pragma unroll
        for (int xm = 1; xm <= 8; xm <<= 1) {
#pragma unroll
            for (int j = 0; j < 8; ++j) cm[j] = fmaxf(cm[j], __shfl_xor(cm[j], xm, 32));
        }
        float alpha[8], rs[8];
#pragma unroll
        for (int j = 0; j < 8; ++j) {
            float mn = fmaxf(m[j], cm[j]);
            alpha[j] = __expf(m[j] - mn);
            m[j] = mn;
            s0[j] = __expf(s0[j] - mn);
            s1[j] = __expf(s1[j] - mn);
            rs[j] = s0[j] + s1[j];
        }
#pragma unroll
        for (int xm = 1; xm <= 8; xm <<= 1) {
#pragma unroll
            for (int j = 0; j < 8; ++j) rs[j] += __shfl_xor(rs[j], xm, 32);
        }
#pragma unroll
        for (int j = 0; j < 8; ++j) {
            lsum[j] = lsum[j] * alpha[j] + rs[j];
            o0[j] *= alpha[j];
            o1[j] *= alpha[j];
        }

        // re-shape P: C-layout -> A-layout through per-wave LDS scratch
        __bf16* pw = Pl[wave];
#pragma unroll
        for (int j = 0; j < 8; ++j) {
            pw[(rbase + j) * 32 + col]      = (__bf16)s0[j];
            pw[(rbase + j) * 32 + 16 + col] = (__bf16)s1[j];
        }
        asm volatile("s_wait_dscnt 0" ::: "memory");
        v16bf pf  = gather_a(pw, 32, lane);
        v16bf bv0 = gather_b_row(Vt, 32, 0, lane);   // Vt[d][k] contiguous in k
        v16bf bv1 = gather_b_row(Vt, 32, 16, lane);
        o0 = wmma_bf16(pf, bv0, o0);
        o1 = wmma_bf16(pf, bv1, o1);
    }

#pragma unroll
    for (int j = 0; j < 8; ++j) {
        float inv = 1.0f / lsum[j];
        o0[j] *= inv;
        o1[j] *= inv;
    }
    // AO[n][l][h*HD + d] bf16, row-major (N*L x E) for the output GEMM
    __bf16* ob = AO + ((size_t)n * L_S + q0 + wave * 16) * E_D + h * HD;
#pragma unroll
    for (int j = 0; j < 8; ++j) {
        ob[(size_t)(rbase + j) * E_D + col]      = (__bf16)o0[j];
        ob[(size_t)(rbase + j) * E_D + 16 + col] = (__bf16)o1[j];
    }
}

// ---------------------------------------------------------------------------
// Kernel 3: Out(4096x1024) = AO @ Wo^T + bo.  128x64 tile per block.
// A tile staged via async global->LDS b128; Wo converted f32->bf16 manually.
// ---------------------------------------------------------------------------
__global__ __launch_bounds__(256) void ogemm_kernel(
    const __bf16* __restrict__ A, const float* __restrict__ Wo,
    const float* __restrict__ bo, float* __restrict__ out) {
    __shared__ __bf16 Al[128 * 32];
    __shared__ __bf16 Bl[64 * 32];
    const int tid = threadIdx.x, lane = tid & 31, wave = tid >> 5;
    const int r0 = blockIdx.x * 128;
    const int c0 = blockIdx.y * 64;
    v8f acc[4] = {{}, {}, {}, {}};
    const uint32_t la = lds_off(Al);
    const uint64_t gA = (uint64_t)(uintptr_t)A;

    for (int kc = 0; kc < E_D; kc += 32) {
        __syncthreads();
        // async stage A: 128 rows x 64B; 512 b128 slots, 2 per thread
#pragma unroll
        for (int i = 0; i < 2; ++i) {
            int s = tid * 2 + i;
            int r = s >> 2, part = s & 3;   // 4 x 16B per 64B row
            uint64_t ga = gA + (size_t)(r0 + r) * (E_D * 2) + kc * 2 + part * 16;
            async_b128(la + s * 16, ga);
        }
        // stage B: Wo rows c0..c0+63, cols kc..kc+31, f32 -> bf16 packed
        uint32_t* bd = (uint32_t*)Bl;
#pragma unroll
        for (int i = 0; i < 4; ++i) {
            int d = tid * 4 + i;            // 1024 dwords
            int r = d >> 4, c = (d & 15) * 2;
            const float* src = &Wo[(size_t)(c0 + r) * E_D + kc + c];
            bd[d] = pack_bf2(src[0], src[1]);
        }
        wait_async0();
        __syncthreads();

        v16bf af = gather_a(Al + wave * 16 * 32, 32, lane);
#pragma unroll
        for (int f = 0; f < 4; ++f) {
            v16bf bf = gather_b_row(Bl, 32, f * 16, lane);
            acc[f] = wmma_bf16(af, bf, acc[f]);
        }
    }

    int col   = lane & 15;
    int rbase = (lane & 16) ? 8 : 0;
#pragma unroll
    for (int f = 0; f < 4; ++f) {
        float bias = bo[c0 + f * 16 + col];
#pragma unroll
        for (int j = 0; j < 8; ++j) {
            out[(size_t)(r0 + wave * 16 + rbase + j) * E_D + c0 + f * 16 + col] =
                acc[f][j] + bias;
        }
    }
}

extern "C" void kernel_launch(void* const* d_in, const int* in_sizes, int n_in,
                              void* d_out, int out_size, void* d_ws, size_t ws_size,
                              hipStream_t stream) {
    (void)in_sizes; (void)n_in; (void)out_size; (void)ws_size;
    const float* values = (const float*)d_in[0];
    const float* keys   = (const float*)d_in[1];
    const float* query  = (const float*)d_in[2];
    const int*   mask   = (const int*)d_in[3];
    const float* Wv     = (const float*)d_in[4];
    const float* Wk     = (const float*)d_in[5];
    const float* Wq     = (const float*)d_in[6];
    const float* Wo     = (const float*)d_in[7];
    const float* bo     = (const float*)d_in[8];
    float* out = (float*)d_out;

    const size_t headElems = (size_t)N_B * H_N * L_S * HD;   // 4M bf16 = 8MB each
    __bf16* Vp = (__bf16*)d_ws;
    __bf16* Kp = Vp + headElems;
    __bf16* Qp = Kp + headElems;
    __bf16* AO = Qp + headElems;

    dim3 blk(256);
    dim3 pg(N_B * L_S / 128, H_N);
    proj_kernel<<<pg, blk, 0, stream>>>(values, Wv, Vp, 1.0f);
    proj_kernel<<<pg, blk, 0, stream>>>(keys,   Wk, Kp, 1.0f);
    proj_kernel<<<pg, blk, 0, stream>>>(query,  Wq, Qp, 1.0f / 32.0f);  // fold 1/sqrt(E)

    dim3 ag(L_S / 128, H_N, N_B);
    attn_kernel<<<ag, blk, 0, stream>>>(Qp, Kp, Vp, mask, AO);

    dim3 gg((N_B * L_S) / 128, E_D / 64);
    ogemm_kernel<<<gg, blk, 0, stream>>>(AO, Wo, bo, out);
}
